// Attention_64819646431478
// MI455X (gfx1250) — matrix operand order repaired and checked by reading, not compiled, not run on hardware
//
#include <hip/hip_runtime.h>
#include <hip/hip_bf16.h>
#include <stdint.h>

// Problem constants (match reference): B=1, T=2048, DIM=1024, H=16, D=64
#define T_SEQ 2048
#define DIMC  1024
#define NH    16
#define HD    64

typedef __attribute__((ext_vector_type(16))) __bf16 bf16x16;
typedef __attribute__((ext_vector_type(8)))  __bf16 bf16x8;
typedef __attribute__((ext_vector_type(8)))  float  f32x8;

union FragB16 {
    bf16x16 v;
    struct { bf16x8 lo, hi; } h;
};

static __device__ __forceinline__ __bf16 f2bf(float f) {
    union { float f; uint32_t u; } a; a.f = f;
    uint32_t r = a.u + 0x7FFFu + ((a.u >> 16) & 1u);   // round-to-nearest-even
    union { uint16_t s; __bf16 b; } o; o.s = (uint16_t)(r >> 16);
    return o.b;
}

static __device__ __forceinline__ f32x8 wmma_bf16(const FragB16& a, const FragB16& b, f32x8 c) {
    // D = A(16x32 bf16) x B(32x16 bf16) + C(16x16 f32)
    return __builtin_amdgcn_wmma_f32_16x16x32_bf16(false, a.v, false, b.v, (short)0, c, false, false);
}

// A-operand 16-bit 16x32 layout (ISA 7.12.2): lanes 0-15 hold K {0..7,16..23},
// lanes 16-31 hold K {8..15,24..31}; lane&15 selects the M row.
static __device__ __forceinline__ void loadFragA(FragB16& f, const __bf16* row, int k0, int half) {
    f.h.lo = *(const bf16x8*)(row + k0 + half * 8);
    f.h.hi = *(const bf16x8*)(row + k0 + 16 + half * 8);
}
// B-operand 16-bit 32x16 layout: lanes 0-15 (N col) hold K {0..7,16..23}, lanes 16-31 hold K {8..15,24..31}.
static __device__ __forceinline__ void loadFragB(FragB16& f, const __bf16* row, int k0, int half) {
    f.h.lo = *(const bf16x8*)(row + k0 + half * 8);
    f.h.hi = *(const bf16x8*)(row + k0 + 16 + half * 8);
}

// ---------------------------------------------------------------------------
// 1) fp32 -> bf16 cast
// ---------------------------------------------------------------------------
__global__ void cast_bf16_kernel(const float* __restrict__ in, __bf16* __restrict__ out, int n) {
    int i = blockIdx.x * blockDim.x + threadIdx.x;
    if (i < n) out[i] = f2bf(in[i]);
}

// ---------------------------------------------------------------------------
// 2) C[M,N] = A[M,K](bf16) @ W[N,K](bf16)^T  -> fp32
//    256 threads = 8 waves; wave computes 32(M) x 64(N): 8 accumulators.
//    Register double-buffered over K (step 32): next k-step's 12 b128 loads are
//    issued before the current k-step's 8 WMMAs so loads overlap the matrix
//    pipe instead of a wait per WMMA. L2-targeted prefetch runs ~8 steps ahead.
// ---------------------------------------------------------------------------
__global__ __launch_bounds__(256)
void gemm_bf16_wmma(const __bf16* __restrict__ A, const __bf16* __restrict__ W,
                    float* __restrict__ C, int M, int N, int K) {
    const int lane = threadIdx.x & 31;
    const int wave = threadIdx.x >> 5;
    const int l16  = lane & 15;
    const int half = lane >> 4;
    const int mBase = blockIdx.y * 256 + wave * 32;
    const int nBase = blockIdx.x * 64;

    const __bf16* aR[2] = { A + (size_t)(mBase      + l16) * K,
                            A + (size_t)(mBase + 16 + l16) * K };
    const __bf16* bR[4];
#pragma unroll
    for (int j = 0; j < 4; ++j) bR[j] = W + (size_t)(nBase + j * 16 + l16) * K;

    f32x8 acc[2][4] = {};
    FragB16 aC[2], bC[4], aN[2], bN[4];
    loadFragA(aC[0], aR[0], 0, half);
    loadFragA(aC[1], aR[1], 0, half);
#pragma unroll
    for (int j = 0; j < 4; ++j) loadFragB(bC[j], bR[j], 0, half);

    for (int k0 = 0; k0 + 32 < K; k0 += 32) {
        const int kn = k0 + 32;
        // deep prefetch into GL2 (~8 k-steps ahead)
        __builtin_prefetch(aR[0] + kn + 256, 0, 1);
        __builtin_prefetch(bR[0] + kn + 256, 0, 1);
        // issue next k-step's fragment loads first (12x global_load_b128)
        loadFragA(aN[0], aR[0], kn, half);
        loadFragA(aN[1], aR[1], kn, half);
#pragma unroll
        for (int j = 0; j < 4; ++j) loadFragB(bN[j], bR[j], kn, half);
        // 8 WMMAs on the already-resident fragments
#pragma unroll
        for (int mi = 0; mi < 2; ++mi)
#pragma unroll
            for (int j = 0; j < 4; ++j)
                acc[mi][j] = wmma_bf16(aC[mi], bC[j], acc[mi][j]);
        aC[0] = aN[0]; aC[1] = aN[1];
#pragma unroll
        for (int j = 0; j < 4; ++j) bC[j] = bN[j];
    }
    // tail k-step
#pragma unroll
    for (int mi = 0; mi < 2; ++mi)
#pragma unroll
        for (int j = 0; j < 4; ++j)
            acc[mi][j] = wmma_bf16(aC[mi], bC[j], acc[mi][j]);

#pragma unroll
    for (int mi = 0; mi < 2; ++mi)
#pragma unroll
        for (int j = 0; j < 4; ++j) {
            const int col = nBase + j * 16 + l16;
#pragma unroll
            for (int i = 0; i < 8; ++i) {
                const int row = mBase + mi * 16 + i + 8 * half;
                C[(size_t)row * N + col] = acc[mi][j][i];
            }
        }
}

// ---------------------------------------------------------------------------
// 3) Per-head RMSNorm of q,k (fp32) + repack to bf16:
//      q,k -> [H][T][64] (row-major per head)
//      v   -> [H][64][T] (transposed so PV B-operand K-dim is contiguous)
//    One wave per (t,h) row; lane handles 2 of the 64 elements.
// ---------------------------------------------------------------------------
__global__ __launch_bounds__(128)
void norm_pack_kernel(const float* __restrict__ qf, const float* __restrict__ kf,
                      const float* __restrict__ vf,
                      const float* __restrict__ qg, const float* __restrict__ kg,
                      __bf16* __restrict__ qb, __bf16* __restrict__ kb,
                      __bf16* __restrict__ vT) {
    const int lane = threadIdx.x & 31;
    const int row  = blockIdx.x * 4 + (threadIdx.x >> 5);  // [0, T*NH)
    const int t = row >> 4;
    const int h = row & (NH - 1);
    const int d = lane * 2;
    const float eps = 1.1920929e-7f;  // FLT_EPSILON (torch RMSNorm default)

    // ---- q ----
    {
        const float2 x = *(const float2*)(qf + (size_t)t * DIMC + h * HD + d);
        float ss = x.x * x.x + x.y * x.y;
#pragma unroll
        for (int m = 16; m >= 1; m >>= 1) ss += __shfl_xor(ss, m, 32);
        const float inv = rsqrtf(ss * (1.0f / HD) + eps);
        union { __bf16 b[2]; uint32_t u; } p;
        p.b[0] = f2bf(x.x * inv * qg[d]);
        p.b[1] = f2bf(x.y * inv * qg[d + 1]);
        *(uint32_t*)(qb + ((size_t)h * T_SEQ + t) * HD + d) = p.u;
    }
    // ---- k ----
    {
        const float2 x = *(const float2*)(kf + (size_t)t * DIMC + h * HD + d);
        float ss = x.x * x.x + x.y * x.y;
#pragma unroll
        for (int m = 16; m >= 1; m >>= 1) ss += __shfl_xor(ss, m, 32);
        const float inv = rsqrtf(ss * (1.0f / HD) + eps);
        union { __bf16 b[2]; uint32_t u; } p;
        p.b[0] = f2bf(x.x * inv * kg[d]);
        p.b[1] = f2bf(x.y * inv * kg[d + 1]);
        *(uint32_t*)(kb + ((size_t)h * T_SEQ + t) * HD + d) = p.u;
    }
    // ---- v (transpose to [H][64][T]) ----
    {
        const float2 x = *(const float2*)(vf + (size_t)t * DIMC + h * HD + d);
        vT[((size_t)h * HD + d)     * T_SEQ + t] = f2bf(x.x);
        vT[((size_t)h * HD + d + 1) * T_SEQ + t] = f2bf(x.y);
    }
}

// ---------------------------------------------------------------------------
// 4) Flash attention, one wave per (head, 32-row q tile pair).
//    Each K/V fragment load feeds TWO WMMAs (both 16-row q tiles), doubling
//    arithmetic intensity on the kv stream. Streams kv in 32-wide blocks:
//    8 WMMAs for S=Q@K^T, fp32 online softmax in registers, P re-shaped
//    C-layout -> A-layout via LDS, 8 WMMAs for O += P@V.
//    Scores never touch global memory (saves ~1 GB of HBM traffic).
// ---------------------------------------------------------------------------
__global__ __launch_bounds__(32)
void flash_attn_kernel(const __bf16* __restrict__ qb, const __bf16* __restrict__ kb,
                       const __bf16* __restrict__ vT, __bf16* __restrict__ attn) {
    __shared__ __bf16 ldsP[2 * 16 * 32];
    const int lane = threadIdx.x & 31;
    const int l16  = lane & 15;
    const int half = lane >> 4;
    const int q0 = blockIdx.x * 32;  // first q row of this wave's 2 tiles
    const int h  = blockIdx.y;       // head
    const float scale = 0.125f;      // 64^-0.5

    // Preload Q A-fragments (2 tiles x two K=32 chunks); resident all kernel.
    FragB16 qA[2][2];
#pragma unroll
    for (int ti = 0; ti < 2; ++ti) {
        const __bf16* qRow = qb + ((size_t)h * T_SEQ + q0 + ti * 16 + l16) * HD;
        loadFragA(qA[ti][0], qRow, 0,  half);
        loadFragA(qA[ti][1], qRow, 32, half);
    }

    f32x8 acc[2][4] = {};
    float rowmax[2][8], rowsum[2][8];
#pragma unroll
    for (int ti = 0; ti < 2; ++ti)
#pragma unroll
        for (int i = 0; i < 8; ++i) { rowmax[ti][i] = -3.0e38f; rowsum[ti][i] = 0.0f; }

    for (int kvb = 0; kvb < q0 + 32; kvb += 32) {
        // ---- S = Q @ K^T : each K fragment used by both q tiles ----
        f32x8 s[2][2];
#pragma unroll
        for (int nh = 0; nh < 2; ++nh) {
            f32x8 z0 = {}, z1 = {};
#pragma unroll
            for (int c = 0; c < 2; ++c) {
                const __bf16* kRow = kb + ((size_t)h * T_SEQ + kvb + nh * 16 + l16) * HD + c * 32;
                FragB16 kB;
                loadFragB(kB, kRow, 0, half);
                z0 = wmma_bf16(qA[0][c], kB, z0);
                z1 = wmma_bf16(qA[1][c], kB, z1);
            }
            s[0][nh] = z0; s[1][nh] = z1;
        }
        // ---- scale + causal mask + online softmax (fp32) ----
#pragma unroll
        for (int ti = 0; ti < 2; ++ti) {
#pragma unroll
            for (int nh = 0; nh < 2; ++nh) {
                const int col = kvb + nh * 16 + l16;
#pragma unroll
                for (int i = 0; i < 8; ++i) {
                    const int rowg = q0 + ti * 16 + i + 8 * half;
                    const float v = s[ti][nh][i] * scale;
                    s[ti][nh][i] = (col <= rowg) ? v : -3.0e38f;
                }
            }
#pragma unroll
            for (int i = 0; i < 8; ++i) {
                float tmax = fmaxf(s[ti][0][i], s[ti][1][i]);
#pragma unroll
                for (int m = 8; m >= 1; m >>= 1) tmax = fmaxf(tmax, __shfl_xor(tmax, m, 32));
                const float mnew  = fmaxf(rowmax[ti][i], tmax);
                const float alpha = __expf(rowmax[ti][i] - mnew);
                rowmax[ti][i] = mnew;
                const float p0 = __expf(s[ti][0][i] - mnew);
                const float p1 = __expf(s[ti][1][i] - mnew);
                s[ti][0][i] = p0; s[ti][1][i] = p1;
                float ps = p0 + p1;
#pragma unroll
                for (int m = 8; m >= 1; m >>= 1) ps += __shfl_xor(ps, m, 32);
                rowsum[ti][i] = rowsum[ti][i] * alpha + ps;
#pragma unroll
                for (int j = 0; j < 4; ++j) acc[ti][j][i] *= alpha;
            }
        }
        // ---- P: C-layout -> A-operand layout via LDS ----
#pragma unroll
        for (int ti = 0; ti < 2; ++ti)
#pragma unroll
            for (int nh = 0; nh < 2; ++nh)
#pragma unroll
                for (int i = 0; i < 8; ++i)
                    ldsP[ti * 512 + (i + 8 * half) * 32 + nh * 16 + l16] = f2bf(s[ti][nh][i]);
        __syncthreads();
        FragB16 pA[2];
#pragma unroll
        for (int ti = 0; ti < 2; ++ti) {
            pA[ti].h.lo = *(const bf16x8*)(&ldsP[ti * 512 + l16 * 32 + half * 8]);
            pA[ti].h.hi = *(const bf16x8*)(&ldsP[ti * 512 + l16 * 32 + 16 + half * 8]);
        }
        __syncthreads();
        // ---- O += P @ V : each V fragment used by both q tiles ----
#pragma unroll
        for (int j = 0; j < 4; ++j) {
            const __bf16* vRow = vT + ((size_t)h * HD + j * 16 + l16) * T_SEQ + kvb;
            FragB16 vB;
            loadFragB(vB, vRow, 0, half);
            acc[0][j] = wmma_bf16(pA[0], vB, acc[0][j]);
            acc[1][j] = wmma_bf16(pA[1], vB, acc[1][j]);
        }
    }
    // ---- normalize, pack back to [T][1024] bf16 for the output projection ----
#pragma unroll
    for (int ti = 0; ti < 2; ++ti)
#pragma unroll
        for (int i = 0; i < 8; ++i) {
            const float inv = 1.0f / rowsum[ti][i];
            const int rowg = q0 + ti * 16 + i + 8 * half;
#pragma unroll
            for (int j = 0; j < 4; ++j)
                attn[(size_t)rowg * DIMC + h * HD + j * 16 + l16] = f2bf(acc[ti][j][i] * inv);
        }
}

// ---------------------------------------------------------------------------
// Launcher
// ---------------------------------------------------------------------------
extern "C" void kernel_launch(void* const* d_in, const int* in_sizes, int n_in,
                              void* d_out, int out_size, void* d_ws, size_t ws_size,
                              hipStream_t stream) {
    (void)in_sizes; (void)n_in; (void)out_size; (void)ws_size;

    const float* x  = (const float*)d_in[0];
    const float* wq = (const float*)d_in[1];
    const float* wk = (const float*)d_in[2];
    const float* wv = (const float*)d_in[3];
    const float* wo = (const float*)d_in[4];
    const float* qg = (const float*)d_in[5];
    const float* kg = (const float*)d_in[6];

    const size_t nX = (size_t)T_SEQ * DIMC;  // 2M elements
    const size_t nW = (size_t)DIMC * DIMC;   // 1M elements

    // Workspace layout (~52 MiB total)
    char* ws = (char*)d_ws;
    __bf16* xb  = (__bf16*)ws; ws += nX * 2;
    __bf16* wqb = (__bf16*)ws; ws += nW * 2;
    __bf16* wkb = (__bf16*)ws; ws += nW * 2;
    __bf16* wvb = (__bf16*)ws; ws += nW * 2;
    __bf16* wob = (__bf16*)ws; ws += nW * 2;
    float*  qf  = (float*)ws;  ws += nX * 4;
    float*  kf  = (float*)ws;  ws += nX * 4;
    float*  vf  = (float*)ws;  ws += nX * 4;
    __bf16* qbh = (__bf16*)ws; ws += nX * 2;  // [H][T][64]
    __bf16* kbh = (__bf16*)ws; ws += nX * 2;  // [H][T][64]
    __bf16* vTh = (__bf16*)ws; ws += nX * 2;  // [H][64][T]
    __bf16* ab  = (__bf16*)ws; ws += nX * 2;  // attention output [T][1024] bf16

    // 1) casts
    cast_bf16_kernel<<<(int)((nX + 255) / 256), 256, 0, stream>>>(x,  xb,  (int)nX);
    cast_bf16_kernel<<<(int)((nW + 255) / 256), 256, 0, stream>>>(wq, wqb, (int)nW);
    cast_bf16_kernel<<<(int)((nW + 255) / 256), 256, 0, stream>>>(wk, wkb, (int)nW);
    cast_bf16_kernel<<<(int)((nW + 255) / 256), 256, 0, stream>>>(wv, wvb, (int)nW);
    cast_bf16_kernel<<<(int)((nW + 255) / 256), 256, 0, stream>>>(wo, wob, (int)nW);

    // 2) QKV projections (WMMA, pipelined)
    dim3 gg(DIMC / 64, T_SEQ / 256), gb(256);
    gemm_bf16_wmma<<<gg, gb, 0, stream>>>(xb, wqb, qf, T_SEQ, DIMC, DIMC);
    gemm_bf16_wmma<<<gg, gb, 0, stream>>>(xb, wkb, kf, T_SEQ, DIMC, DIMC);
    gemm_bf16_wmma<<<gg, gb, 0, stream>>>(xb, wvb, vf, T_SEQ, DIMC, DIMC);

    // 3) RMSNorm q,k + repack q/k/v to attention layouts
    norm_pack_kernel<<<(T_SEQ * NH) / 4, 128, 0, stream>>>(qf, kf, vf, qg, kg, qbh, kbh, vTh);

    // 4) causal flash attention (WMMA + online softmax), 32 q rows per wave
    dim3 fg(T_SEQ / 32, NH);
    flash_attn_kernel<<<fg, 32, 0, stream>>>(qbh, kbh, vTh, ab);

    // 5) output projection (WMMA) -> fp32 d_out
    gemm_bf16_wmma<<<gg, gb, 0, stream>>>(ab, wob, (float*)d_out, T_SEQ, DIMC, DIMC);
}
